// E3Critic_82764019794074
// MI455X (gfx1250) — matrix operand order repaired and checked by reading, not compiled, not run on hardware
//
#include <hip/hip_runtime.h>
#include <hip/hip_bf16.h>

#define NUM_AGENTS 32
#define N_OBS      16
#define KNN        5
#define NNODE      80
#define E_KNN      400           // 80*5
#define NEDGE      416           // + 16 agent->goal edges
#define HID        128
#define HSTR       136           // f16 activation row stride (16B aligned rows)
#define NEG_HUGE  (-3.402823466e38f)
#define FLT_BIG    (3.402823466e38f)

typedef __attribute__((ext_vector_type(16))) _Float16 v16h;
typedef __attribute__((ext_vector_type(8)))  float    v8f;

// out[80][128] (LDS, f32) = h[80][K] (LDS, f16, stride HSTR) @ Wg[K][128] (global f32) + bias
// One wave owns one 16-wide column tile (wave == nt, 8 waves == 128 cols).
// B-fragment loaded once per k-step and reused across the 5 row tiles.
template<int K, int KPAD>
__device__ __forceinline__ void gemm_tile(const _Float16* __restrict__ h,
                                          const float* __restrict__ Wg,
                                          const float* __restrict__ bsh,
                                          float* __restrict__ outp,
                                          int lane, int wave) {
    const int half = lane >> 4;
    const int lm   = lane & 15;
    const int n    = wave * 16 + lm;
    const v8f vzero = {0.f, 0.f, 0.f, 0.f, 0.f, 0.f, 0.f, 0.f};
    v8f acc[5];
#pragma unroll
    for (int mt = 0; mt < 5; ++mt) acc[mt] = vzero;
#pragma unroll
    for (int k0 = 0; k0 < KPAD; k0 += 32) {
        // B 32x16 fragment: lane half selects K block of 16, n = lane&15
        v16h bfrag;
#pragma unroll
        for (int i = 0; i < 16; ++i) {
            int k = k0 + half * 16 + i;
            bfrag[i] = (_Float16)((k < K) ? Wg[k * HID + n] : 0.0f);
        }
#pragma unroll
        for (int mt = 0; mt < 5; ++mt) {
            // A 16x32 fragment: lane half selects K block of 16 (k = 16*half + i), matching B
            const _Float16* hr = h + (mt * 16 + lm) * HSTR + k0;
            v16h afrag;
#pragma unroll
            for (int i = 0; i < 8; ++i) {
                afrag[i]     = hr[half * 16 + i];
                afrag[8 + i] = hr[half * 16 + 8 + i];
            }
            acc[mt] = __builtin_amdgcn_wmma_f32_16x16x32_f16(
                false, afrag, false, bfrag, (short)0, acc[mt], false, false);
        }
    }
    const float bv = bsh[n];
#pragma unroll
    for (int mt = 0; mt < 5; ++mt) {
#pragma unroll
        for (int v = 0; v < 8; ++v) {
            int row = mt * 16 + v + half * 8;   // C/D layout: VGPR v -> M = v (+8 for hi half)
            outp[row * HID + n] = acc[mt][v] + bv;
        }
    }
}

// One full GATv2 hidden layer (F_out = 128), everything in LDS.
template<int K, int KPAD, bool RELU>
__device__ __forceinline__ void gat_layer(
    const float* __restrict__ Wl,  const float* __restrict__ bl,
    const float* __restrict__ Wr,  const float* __restrict__ br,
    const float* __restrict__ We,  const float* __restrict__ att,
    const float* __restrict__ bias,
    _Float16* s_h, float* s_xl, float* s_xr,
    float* s_we, float* s_att, float* s_bl, float* s_br, float* s_bias,
    const int* s_src, const int* s_dup, const float* s_ea, float* s_logit,
    int tid, int lane, int wave) {
    // stage per-layer vectors
    for (int i = tid; i < HID; i += 256) {
        s_att[i] = att[i]; s_bl[i] = bl[i]; s_br[i] = br[i]; s_bias[i] = bias[i];
    }
    for (int i = tid; i < 5 * HID; i += 256) s_we[i] = We[i];
    __syncthreads();

    // xl = h@Wl + bl ; xr = h@Wr + br   (WMMA)
    gemm_tile<K, KPAD>(s_h, Wl, s_bl, s_xl, lane, wave);
    gemm_tile<K, KPAD>(s_h, Wr, s_br, s_xr, lane, wave);
    __syncthreads();

    // edge logits: wave per edge, 4 features per lane, shfl reduction
    for (int e = wave; e < NEDGE; e += 8) {
        const int src = s_src[e];
        const int dst = (e < E_KNN) ? (e / 5) : (e - E_KNN + NUM_AGENTS);
        const float e0 = s_ea[e * 5 + 0], e1 = s_ea[e * 5 + 1], e2 = s_ea[e * 5 + 2];
        const float e3 = s_ea[e * 5 + 3], e4 = s_ea[e * 5 + 4];
        float part = 0.f;
#pragma unroll
        for (int q = 0; q < 4; ++q) {
            int f = lane + 32 * q;
            float ew = e0 * s_we[0 * HID + f] + e1 * s_we[1 * HID + f] +
                       e2 * s_we[2 * HID + f] + e3 * s_we[3 * HID + f] +
                       e4 * s_we[4 * HID + f];
            float t = s_xl[src * HID + f] + s_xr[dst * HID + f] + ew;
            t = (t > 0.f) ? t : 0.2f * t;          // leaky_relu(0.2)
            part += t * s_att[f];
        }
#pragma unroll
        for (int off = 16; off > 0; off >>= 1) part += __shfl_xor(part, off, 32);
        if (lane == 0) {
            bool masked = (e >= E_KNN) && s_dup[e - E_KNN];
            s_logit[e] = masked ? NEG_HUGE : part;
        }
    }
    __syncthreads();

    // segment softmax + aggregation: wave per dst node; lanes span features
    for (int n = wave; n < NNODE; n += 8) {
        const int base = 5 * n;
        const float l0 = s_logit[base + 0], l1 = s_logit[base + 1], l2 = s_logit[base + 2];
        const float l3 = s_logit[base + 3], l4 = s_logit[base + 4];
        const bool hasExtra = (n >= NUM_AGENTS) && (n < NUM_AGENTS + N_OBS);
        const int  esrc = hasExtra ? (n - NUM_AGENTS) : 0;
        const float lex = hasExtra ? s_logit[E_KNN + (n - NUM_AGENTS)] : NEG_HUGE;
        float m = fmaxf(fmaxf(fmaxf(l0, l1), fmaxf(l2, l3)), fmaxf(l4, lex));
        const float a0 = expf(l0 - m), a1 = expf(l1 - m), a2 = expf(l2 - m);
        const float a3 = expf(l3 - m), a4 = expf(l4 - m);
        const float aex = hasExtra ? expf(lex - m) : 0.f;
        const float inv = 1.f / (a0 + a1 + a2 + a3 + a4 + aex);
        const int s0 = s_src[base + 0], s1 = s_src[base + 1], s2 = s_src[base + 2];
        const int s3 = s_src[base + 3], s4 = s_src[base + 4];
        for (int f = lane; f < HID; f += 32) {
            float acc = a0 * s_xl[s0 * HID + f] + a1 * s_xl[s1 * HID + f] +
                        a2 * s_xl[s2 * HID + f] + a3 * s_xl[s3 * HID + f] +
                        a4 * s_xl[s4 * HID + f] + aex * s_xl[esrc * HID + f];
            float o = acc * inv + s_bias[f];
            if (RELU) o = fmaxf(o, 0.f);
            s_h[n * HSTR + f] = (_Float16)o;
        }
    }
    __syncthreads();
}

__global__ __launch_bounds__(256) void e3critic_kernel(
    const float* __restrict__ obst, const float* __restrict__ apos,
    const float* __restrict__ gpos, const float* __restrict__ avel,
    const float* __restrict__ Wl1, const float* __restrict__ bl1,
    const float* __restrict__ Wr1, const float* __restrict__ br1,
    const float* __restrict__ We1, const float* __restrict__ att1,
    const float* __restrict__ bias1,
    const float* __restrict__ Wl2, const float* __restrict__ bl2,
    const float* __restrict__ Wr2, const float* __restrict__ br2,
    const float* __restrict__ We2, const float* __restrict__ att2,
    const float* __restrict__ bias2,
    const float* __restrict__ Wl3, const float* __restrict__ bl3,
    const float* __restrict__ Wr3, const float* __restrict__ br3,
    const float* __restrict__ We3, const float* __restrict__ att3,
    const float* __restrict__ bias3,
    float* __restrict__ outp) {
    __shared__ float    s_pos[NNODE][2];
    __shared__ float    s_vel[NNODE][2];
    __shared__ float    s_rad[NNODE];
    __shared__ int      s_nbr[NNODE][KNN];
    __shared__ int      s_dup[N_OBS];
    __shared__ int      s_src[NEDGE];
    __shared__ float    s_ea[NEDGE * 5];
    __shared__ float    s_logit[NEDGE];
    __shared__ _Float16 s_h[NNODE * HSTR];
    __shared__ float    s_xl[NNODE * HID];
    __shared__ float    s_xr[NNODE * HID];
    __shared__ float    s_we[5 * HID];
    __shared__ float    s_att[HID], s_bl[HID], s_br[HID], s_bias[HID];
    __shared__ float    s_xl3[NNODE], s_xr3[NNODE];
    __shared__ float    s_out3[NUM_AGENTS];
    __shared__ float    s_sum;

    const int tid  = threadIdx.x;
    const int lane = tid & 31;
    const int wave = tid >> 5;
    const int b    = blockIdx.x;

    // warm L2 for the big weights (global_prefetch_b8)
    for (int i = tid * 32; i < HID * HID; i += 256 * 32) {
        __builtin_prefetch(Wl2 + i, 0, 3);
        __builtin_prefetch(Wr2 + i, 0, 3);
    }

    // ---- stage graph inputs: pos = [agents | goals | obstacles] ----
    for (int n = tid; n < NNODE; n += 256) {
        float px, py, vx = 0.f, vy = 0.f, r = 0.f;
        if (n < NUM_AGENTS) {
            px = apos[(b * NUM_AGENTS + n) * 2 + 0];
            py = apos[(b * NUM_AGENTS + n) * 2 + 1];
            vx = avel[(b * NUM_AGENTS + n) * 2 + 0];
            vy = avel[(b * NUM_AGENTS + n) * 2 + 1];
            r  = 0.05f;
        } else if (n < 2 * NUM_AGENTS) {
            int g = n - NUM_AGENTS;
            px = gpos[(b * NUM_AGENTS + g) * 2 + 0];
            py = gpos[(b * NUM_AGENTS + g) * 2 + 1];
        } else {
            int o = n - 2 * NUM_AGENTS;
            px = obst[(b * N_OBS + o) * 2 + 0];
            py = obst[(b * N_OBS + o) * 2 + 1];
            r  = 0.1f;
        }
        s_pos[n][0] = px; s_pos[n][1] = py;
        s_vel[n][0] = vx; s_vel[n][1] = vy;
        s_rad[n] = r;
    }
    __syncthreads();

    // ---- 5-NN per node (incl self); strict < preserves top_k tie order ----
    if (tid < NNODE) {
        float bd[KNN]; int bi[KNN];
#pragma unroll
        for (int s = 0; s < KNN; ++s) { bd[s] = FLT_BIG; bi[s] = 0; }
        const float pix = s_pos[tid][0], piy = s_pos[tid][1];
        for (int j = 0; j < NNODE; ++j) {
            float dx = pix - s_pos[j][0], dy = piy - s_pos[j][1];
            float d2 = dx * dx + dy * dy;
            if (d2 < bd[KNN - 1]) {
                bd[KNN - 1] = d2; bi[KNN - 1] = j;
#pragma unroll
                for (int s = KNN - 1; s > 0; --s) {
                    if (bd[s] < bd[s - 1]) {
                        float td = bd[s]; bd[s] = bd[s - 1]; bd[s - 1] = td;
                        int   ti = bi[s]; bi[s] = bi[s - 1]; bi[s - 1] = ti;
                    }
                }
            }
        }
#pragma unroll
        for (int s = 0; s < KNN; ++s) s_nbr[tid][s] = bi[s];
    }
    __syncthreads();

    // ---- edge lists, dedup mask, node features ----
    if (tid < N_OBS) {
        int d = 0;
#pragma unroll
        for (int s = 0; s < KNN; ++s) d |= (s_nbr[NUM_AGENTS + tid][s] == tid);
        s_dup[tid] = d;
    }
    for (int e = tid; e < NEDGE; e += 256)
        s_src[e] = (e < E_KNN) ? s_nbr[e / 5][e % 5] : (e - E_KNN);
    for (int idx = tid; idx < NNODE * 32; idx += 256) {   // x padded to K=32
        int n = idx >> 5, c = idx & 31;
        float v = 0.f;
        if (c == 0)      v = (n < NUM_AGENTS) ? 1.f : 0.f;
        else if (c == 1) v = (n >= NUM_AGENTS && n < 2 * NUM_AGENTS) ? 1.f : 0.f;
        else if (c == 2) v = (n >= 2 * NUM_AGENTS) ? 1.f : 0.f;
        else if (c == 3) v = (n < NUM_AGENTS)
                           ? sqrtf(s_vel[n][0] * s_vel[n][0] + s_vel[n][1] * s_vel[n][1]) : 0.f;
        else if (c == 4) v = s_rad[n];
        s_h[n * HSTR + c] = (_Float16)v;
    }
    __syncthreads();

    // ---- edge attributes ----
    for (int e = tid; e < NEDGE; e += 256) {
        const int src = s_src[e];
        const int dst = (e < E_KNN) ? (e / 5) : (e - E_KNN + NUM_AGENTS);
        float pdx = s_pos[src][0] - s_pos[dst][0];
        float pdy = s_pos[src][1] - s_pos[dst][1];
        float dist = sqrtf(pdx * pdx + pdy * pdy);
        float inv  = 1.f / fmaxf(dist, 1e-6f);
        float pnx = pdx * inv, pny = pdy * inv;
        float rvx = s_vel[src][0] - s_vel[dst][0];
        float rvy = s_vel[src][1] - s_vel[dst][1];
        s_ea[e * 5 + 0] = ((src < NUM_AGENTS) && (dst == src + NUM_AGENTS)) ? 1.f : 0.f;
        s_ea[e * 5 + 1] = dist;
        s_ea[e * 5 + 2] = dist - (s_rad[src] + s_rad[dst]);
        s_ea[e * 5 + 3] = rvx * pnx + rvy * pny;
        s_ea[e * 5 + 4] = rvx * pny - rvy * pnx;
    }
    __syncthreads();

    // ---- layers 1 & 2 (WMMA GEMMs + softmax aggregation) ----
    gat_layer<5, 32, true>(Wl1, bl1, Wr1, br1, We1, att1, bias1,
                           s_h, s_xl, s_xr, s_we, s_att, s_bl, s_br, s_bias,
                           s_src, s_dup, s_ea, s_logit, tid, lane, wave);
    gat_layer<HID, HID, true>(Wl2, bl2, Wr2, br2, We2, att2, bias2,
                              s_h, s_xl, s_xr, s_we, s_att, s_bl, s_br, s_bias,
                              s_src, s_dup, s_ea, s_logit, tid, lane, wave);

    // ---- layer 3 (F_out = 1), scalar VALU path ----
    if (tid < NNODE) {
        float xlv = bl3[0], xrv = br3[0];
        for (int k = 0; k < HID; ++k) {
            float hv = (float)s_h[tid * HSTR + k];
            xlv += hv * Wl3[k];
            xrv += hv * Wr3[k];
        }
        s_xl3[tid] = xlv; s_xr3[tid] = xrv;
    }
    __syncthreads();
    for (int e = tid; e < NEDGE; e += 256) {
        const int src = s_src[e];
        const int dst = (e < E_KNN) ? (e / 5) : (e - E_KNN + NUM_AGENTS);
        float ew = s_ea[e * 5 + 0] * We3[0] + s_ea[e * 5 + 1] * We3[1] +
                   s_ea[e * 5 + 2] * We3[2] + s_ea[e * 5 + 3] * We3[3] +
                   s_ea[e * 5 + 4] * We3[4];
        float t = s_xl3[src] + s_xr3[dst] + ew;
        t = (t > 0.f) ? t : 0.2f * t;
        bool masked = (e >= E_KNN) && s_dup[e - E_KNN];
        s_logit[e] = masked ? NEG_HUGE : t * att3[0];
    }
    __syncthreads();
    // only agent nodes (0..31) feed the pool; they each have exactly 5 knn edges
    if (tid < NUM_AGENTS) {
        const int base = 5 * tid;
        float l0 = s_logit[base], l1 = s_logit[base + 1], l2 = s_logit[base + 2];
        float l3 = s_logit[base + 3], l4 = s_logit[base + 4];
        float m = fmaxf(fmaxf(fmaxf(l0, l1), fmaxf(l2, l3)), l4);
        float a0 = expf(l0 - m), a1 = expf(l1 - m), a2 = expf(l2 - m);
        float a3 = expf(l3 - m), a4 = expf(l4 - m);
        float inv = 1.f / (a0 + a1 + a2 + a3 + a4);
        float o = (a0 * s_xl3[s_src[base + 0]] + a1 * s_xl3[s_src[base + 1]] +
                   a2 * s_xl3[s_src[base + 2]] + a3 * s_xl3[s_src[base + 3]] +
                   a4 * s_xl3[s_src[base + 4]]) * inv + bias3[0];
        s_out3[tid] = o;
    }
    __syncthreads();
    if (tid == 0) {
        float s = 0.f;
        for (int n = 0; n < NUM_AGENTS; ++n) s += s_out3[n];
        s_sum = s;
    }
    __syncthreads();
    if (tid < NUM_AGENTS) outp[b * NUM_AGENTS + tid] = s_sum;
}

extern "C" void kernel_launch(void* const* d_in, const int* in_sizes, int n_in,
                              void* d_out, int out_size, void* d_ws, size_t ws_size,
                              hipStream_t stream) {
    const float* obst  = (const float*)d_in[0];
    const float* apos  = (const float*)d_in[1];
    const float* gpos  = (const float*)d_in[2];
    const float* avel  = (const float*)d_in[3];
    const float* Wl1   = (const float*)d_in[4];
    const float* bl1   = (const float*)d_in[5];
    const float* Wr1   = (const float*)d_in[6];
    const float* br1   = (const float*)d_in[7];
    const float* We1   = (const float*)d_in[8];
    const float* att1  = (const float*)d_in[9];
    const float* bias1 = (const float*)d_in[10];
    const float* Wl2   = (const float*)d_in[11];
    const float* bl2   = (const float*)d_in[12];
    const float* Wr2   = (const float*)d_in[13];
    const float* br2   = (const float*)d_in[14];
    const float* We2   = (const float*)d_in[15];
    const float* att2  = (const float*)d_in[16];
    const float* bias2 = (const float*)d_in[17];
    const float* Wl3   = (const float*)d_in[18];
    const float* bl3   = (const float*)d_in[19];
    const float* Wr3   = (const float*)d_in[20];
    const float* br3   = (const float*)d_in[21];
    const float* We3   = (const float*)d_in[22];
    const float* att3  = (const float*)d_in[23];
    const float* bias3 = (const float*)d_in[24];
    float* outp = (float*)d_out;

    const int B = in_sizes[0] / (N_OBS * 2);   // 1024
    e3critic_kernel<<<B, 256, 0, stream>>>(
        obst, apos, gpos, avel,
        Wl1, bl1, Wr1, br1, We1, att1, bias1,
        Wl2, bl2, Wr2, br2, We2, att2, bias2,
        Wl3, bl3, Wr3, br3, We3, att3, bias3,
        outp);
}
